// ArcFaceLoss_40630390620832
// MI455X (gfx1250) — matrix operand order repaired and checked by reading, not compiled, not run on hardware
//
#include <hip/hip_runtime.h>
#include <math.h>

// ---------------- problem constants ----------------
#define BATCH 512
#define DIM   512
#define NCLS  100000
#define BM    128            // M rows per pass (8 waves x 16)
#define BN    128            // classes per block
#define BK    32             // K chunk per WMMA step
#define LPAD  16             // bf16 row padding (32B) to stagger LDS banks
#define NBLK  ((NCLS + BN - 1) / BN)   // 782 class blocks
#define NMB   (BATCH / BM)   // 4 M passes per block

#define SSCALE 120.0f
#define COS_M  0.9553364891256060f     // cos(0.3)
#define SIN_M  0.2955202066613396f     // sin(0.3)
#define TH_    (-0.9553364891256060f)  // cos(pi - 0.3)
#define MM_    (0.2955202066613396f * 0.3f) // sin(pi-m)*m

typedef __bf16 bf16;
typedef __attribute__((ext_vector_type(16))) __bf16 v16bf;
typedef __attribute__((ext_vector_type(8)))  __bf16 v8bf;
typedef __attribute__((ext_vector_type(8)))  float  v8f;

// ---------------------------------------------------------------------------
// Kernel A: L2-normalize x rows; emit fp32 copy (exact target dot) and bf16
// copy (WMMA A operand).
// ---------------------------------------------------------------------------
__global__ __launch_bounds__(128) void k_normalize(
    const float* __restrict__ x, float* __restrict__ xf, bf16* __restrict__ xb) {
  __shared__ float red[128];
  const int b = blockIdx.x, t = threadIdx.x;
  const float4 v = ((const float4*)(x + (long)b * DIM))[t];
  red[t] = v.x*v.x + v.y*v.y + v.z*v.z + v.w*v.w;
  __syncthreads();
  for (int s = 64; s > 0; s >>= 1) { if (t < s) red[t] += red[t + s]; __syncthreads(); }
  const float inv = 1.0f / fmaxf(sqrtf(red[0]), 1e-12f);
  float4 o; o.x = v.x*inv; o.y = v.y*inv; o.z = v.z*inv; o.w = v.w*inv;
  ((float4*)(xf + (long)b * DIM))[t] = o;
  bf16* dst = xb + (long)b * DIM + t * 4;
  dst[0] = (bf16)o.x; dst[1] = (bf16)o.y; dst[2] = (bf16)o.z; dst[3] = (bf16)o.w;
}

// ---------------------------------------------------------------------------
// Kernel B: fused bf16-WMMA GEMM + W row-norms + online-softmax partials.
// One block per 128-class block (grid = NBLK).  W is streamed from HBM
// EXACTLY ONCE into a persistent full-K LDS tile (132 KB) during the first
// M pass; passes 1..3 run the K loop purely out of LDS with no barriers.
// 256 threads = 8 wave32; each wave owns a 16(M) x 128(N) strip per pass.
// ---------------------------------------------------------------------------
__global__ __launch_bounds__(256) void k_gemm_softmax(
    const float* __restrict__ W, const bf16* __restrict__ Xb,
    float2* __restrict__ partials) {
  __shared__ __align__(32) bf16 ldsB[BN][DIM + LPAD];  // 135168 B, persistent W tile
  __shared__ float sqbuf[256];
  __shared__ float invn[BN];

  const int tid  = threadIdx.x;
  const int lane = tid & 31;
  const int wv   = tid >> 5;           // wave id 0..7
  const int cls0 = blockIdx.x * BN;
  const int nloc = lane & 15;
  const int half = lane >> 4;

  // staging role: thread t loads 16 contiguous fp32 of class (t>>1), k-half (t&1)
  const int sCls = tid >> 1;
  const int sK   = (tid & 1) * 16;
  int gc = cls0 + sCls; if (gc >= NCLS) gc = NCLS - 1;   // clamp (masked later)
  const float* wrow = W + (long)gc * DIM;

  float sq = 0.0f;                                       // fused sum(w^2)

  auto stage = [&](int kc) {
    const float4* src = (const float4*)(wrow + kc + sK);
#pragma unroll
    for (int j = 0; j < 4; ++j) {
      const float4 v = src[j];
      sq += v.x*v.x + v.y*v.y + v.z*v.z + v.w*v.w;
      bf16* d = &ldsB[sCls][kc + sK + j * 4];
      d[0] = (bf16)v.x; d[1] = (bf16)v.y; d[2] = (bf16)v.z; d[3] = (bf16)v.w;
    }
  };

  auto zero_acc = [](v8f* acc) {
#pragma unroll
    for (int nt = 0; nt < 8; ++nt)
#pragma unroll
      for (int j = 0; j < 8; ++j) acc[nt][j] = 0.0f;
  };

  // 8 WMMAs over one K-chunk for this wave's current 16 rows
  auto kstep = [&](v8f* acc, const bf16* xrow, int kc) {
    // A frag: lane holds row M=lane%16;
    // lanes<16: K {0..15}; lanes>=16: K {16..31} (matches B).
    const v16bf a = *(const v16bf*)(xrow + kc + half * 16);
#pragma unroll
    for (int nt = 0; nt < 8; ++nt) {
      // B frag: lane holds column N=lane%16, 16 contiguous K values
      const v16bf bfrag = *(const v16bf*)&ldsB[nt * 16 + nloc][kc + half * 16];
      acc[nt] = __builtin_amdgcn_wmma_f32_16x16x32_bf16(
          false, a, false, bfrag, (short)0, acc[nt], false, false);
    }
  };

  // per-row block-local softmax partials: (max, sum exp(l - max)).
  // C/D layout: VGPR r, lanes<16 -> M=r, lanes>=16 -> M=r+8; N = lane%16.
  auto emit_partials = [&](v8f* acc, int m0) {
#pragma unroll
    for (int r = 0; r < 8; ++r) {
      float m = -1e30f;
      float lg[8];
#pragma unroll
      for (int nt = 0; nt < 8; ++nt) {
        const int n = nt * 16 + nloc;
        float l = SSCALE * (acc[nt][r] * invn[n]);
        if (cls0 + n >= NCLS) l = -1e30f;   // mask tail classes
        lg[nt] = l;
        m = fmaxf(m, l);
      }
#pragma unroll
      for (int off = 1; off < 16; off <<= 1) m = fmaxf(m, __shfl_xor(m, off, 32));
      float s = 0.0f;
#pragma unroll
      for (int nt = 0; nt < 8; ++nt) s += __expf(lg[nt] - m);
#pragma unroll
      for (int off = 1; off < 16; off <<= 1) s += __shfl_xor(s, off, 32);
      if (nloc == 0) {
        const int row = m0 + half * 8 + r;
        partials[(long)row * NBLK + blockIdx.x] = make_float2(m, s);
      }
    }
  };

  const int NK = DIM / BK;   // 16 K-chunks
  v8f acc[8];
  zero_acc(acc);

  // ---- M pass 0: pipelined HBM->LDS staging (W read once) + WMMA ----
  stage(0);
  __syncthreads();
  {
    const bf16* xrow = Xb + (long)(wv * 16 + nloc) * DIM;
    for (int kci = 0; kci < NK; ++kci) {
      const int kc = kci * BK;
      if (kci + 1 < NK) {
        if (kci + 2 < NK) __builtin_prefetch(wrow + kc + 2 * BK + sK, 0, 0);
        stage(kc + BK);
      }
      kstep(acc, xrow, kc);
      __syncthreads();
    }
  }

  // finish W row inverse norms (sq complete after last stage)
  sqbuf[tid] = sq;
  __syncthreads();
  if (tid < BN) {
    const float w2 = sqbuf[2 * tid] + sqbuf[2 * tid + 1];
    invn[tid] = rsqrtf(fmaxf(w2, 1e-24f));
  }
  __syncthreads();

  emit_partials(acc, wv * 16);

  // ---- M passes 1..3: W tile is LDS-resident, no barriers needed ----
  for (int mb = 1; mb < NMB; ++mb) {
    zero_acc(acc);
    const int m0 = mb * BM + wv * 16;
    const bf16* xrow = Xb + (long)(m0 + nloc) * DIM;
    for (int kci = 0; kci < NK; ++kci) kstep(acc, xrow, kci * BK);
    emit_partials(acc, m0);
  }
}

// ---------------------------------------------------------------------------
// Kernel C: per batch row — exact fp32 target cosine + margin, merge the 782
// softmax partials (online logsumexp), emit nll[b].  One block per row.
// ---------------------------------------------------------------------------
__global__ __launch_bounds__(256) void k_finalize(
    const float* __restrict__ W, const float* __restrict__ Xf,
    const int* __restrict__ label, const float2* __restrict__ partials,
    float* __restrict__ nll) {
  __shared__ float ra[256], rb[256];
  const int b = blockIdx.x, t = threadIdx.x;
  const long lbl = (long)label[b];
  const float* wrow = W + lbl * (long)DIM;
  const float* xrow = Xf + (long)b * DIM;

  float dot = 0.f, w2 = 0.f;
  for (int k = t; k < DIM; k += 256) {
    const float wvv = wrow[k];
    dot += wvv * xrow[k];
    w2  += wvv * wvv;
  }
  ra[t] = dot; rb[t] = w2; __syncthreads();
  for (int s = 128; s > 0; s >>= 1) {
    if (t < s) { ra[t] += ra[t + s]; rb[t] += rb[t + s]; }
    __syncthreads();
  }
  const float dotAll = ra[0];
  const float w2All  = rb[0];
  __syncthreads();

  // merge (max, sum) partials across NBLK class blocks
  float m = -1e30f, s = 0.f;
  for (int i = t; i < NBLK; i += 256) {
    const float2 p = partials[(long)b * NBLK + i];
    const float nm = fmaxf(m, p.x);
    s = s * __expf(m - nm) + p.y * __expf(p.x - nm);
    m = nm;
  }
  ra[t] = m; rb[t] = s; __syncthreads();
  for (int st = 128; st > 0; st >>= 1) {
    if (t < st) {
      const float m2 = ra[t + st], s2 = rb[t + st];
      const float nm = fmaxf(ra[t], m2);
      rb[t] = rb[t] * __expf(ra[t] - nm) + s2 * __expf(m2 - nm);
      ra[t] = nm;
    }
    __syncthreads();
  }

  if (t == 0) {
    const float M = ra[0], T = rb[0];
    const float cosv = dotAll * rsqrtf(fmaxf(w2All, 1e-24f));
    const float sinv = sqrtf(fmaxf(1.f - cosv * cosv, 0.f));
    float phi = cosv * COS_M - sinv * SIN_M;
    phi = (cosv > TH_) ? phi : (cosv - MM_);
    const float tl = SSCALE * phi;    // margin logit at label
    const float cl = SSCALE * cosv;   // plain logit at label (already in T)
    const float T2 = T + __expf(tl - M) - __expf(cl - M);
    nll[b] = (M + logf(T2)) - tl;
  }
}

// ---------------------------------------------------------------------------
// Kernel D: mean over 512 rows -> scalar (deterministic, no atomics).
// ---------------------------------------------------------------------------
__global__ __launch_bounds__(256) void k_mean(
    const float* __restrict__ nll, float* __restrict__ out) {
  __shared__ float red[256];
  const int t = threadIdx.x;
  red[t] = nll[t] + nll[t + 256];
  __syncthreads();
  for (int s = 128; s > 0; s >>= 1) { if (t < s) red[t] += red[t + s]; __syncthreads(); }
  if (t == 0) out[0] = red[0] * (1.0f / BATCH);
}

// ---------------------------------------------------------------------------
extern "C" void kernel_launch(void* const* d_in, const int* in_sizes, int n_in,
                              void* d_out, int out_size, void* d_ws, size_t ws_size,
                              hipStream_t stream) {
  const float* x     = (const float*)d_in[0];
  const int*   label = (const int*)d_in[1];
  const float* W     = (const float*)d_in[2];
  float*       out   = (float*)d_out;

  char* ws = (char*)d_ws;
  bf16*   xb       = (bf16*)ws;                                   // 512 KB
  float*  xf       = (float*)(ws + (size_t)512 * 1024);           // 1 MB
  float2* partials = (float2*)(ws + (size_t)512 * 1024 + (size_t)1024 * 1024);
  float*  nll      = (float*)((char*)partials + (size_t)BATCH * NBLK * sizeof(float2));

  k_normalize<<<BATCH, 128, 0, stream>>>(x, xf, xb);
  k_gemm_softmax<<<NBLK, 256, 0, stream>>>(W, xb, partials);
  k_finalize<<<BATCH, 256, 0, stream>>>(W, xf, label, partials, nll);
  k_mean<<<1, 256, 0, stream>>>(nll, out);
}
